// _BasicBlock_36034775613690
// MI455X (gfx1250) — matrix operand order repaired and checked by reading, not compiled, not run on hardware
//
#include <hip/hip_runtime.h>
#include <stdint.h>

// Problem constants (fixed by setup_inputs)
#define BATCH 8
#define CHN   128
#define HT    128
#define WD    128
#define HWSZ  (HT*WD)        // 16384
#define NGROUP 16
#define CG    (CHN/NGROUP)   // 8
#define EPSV  1e-5f

typedef __attribute__((ext_vector_type(16))) __bf16 v16bf;
typedef __attribute__((ext_vector_type(8)))  float  v8f;

union ABPack { uint4 u[2]; v16bf v; };

// ---------------- reduction helper ----------------
__device__ inline float waveReduce(float v) {
  #pragma unroll
  for (int o = 16; o > 0; o >>= 1) v += __shfl_down(v, o, 32);
  return v;
}

// ---------------- tiny init ----------------
__global__ void k_zero(float* __restrict__ p, int n) {
  int i = blockIdx.x * 256 + threadIdx.x;
  if (i < n) p[i] = 0.f;
}

// ---------------- weight transform: OIHW f32 -> [tap][co][ci] bf16 ----------------
__global__ void k_wconv(const float* __restrict__ w, __bf16* __restrict__ wtb) {
  int i = blockIdx.x * 256 + threadIdx.x;
  if (i >= 9 * CHN * CHN) return;
  int ci  = i & (CHN - 1);
  int co  = (i >> 7) & (CHN - 1);
  int tap = i >> 14;
  wtb[i] = (__bf16)w[(co * CHN + ci) * 9 + tap];
}

// ---------------- x*mask -> bf16 NHWC staging ----------------
__global__ void k_prep(const float* __restrict__ x, const uint8_t* __restrict__ mask,
                       __bf16* __restrict__ xb) {
  int i = blockIdx.x * 256 + threadIdx.x;          // over B*C*HW/4
  if (i >= BATCH * CHN * HWSZ / 4) return;
  int p4 = i & (HWSZ / 4 - 1);                     // 4096 = 2^12
  int c  = (i >> 12) & (CHN - 1);
  int b  = i >> 19;
  float4 v = ((const float4*)x)[(size_t)i];        // NCHW coalesced read
  const uint8_t* mp = mask + b * HWSZ + p4 * 4;
  int base = (b * HWSZ + p4 * 4) * CHN + c;        // NHWC write
  xb[base]           = (__bf16)(v.x * (float)mp[0]);
  xb[base + CHN]     = (__bf16)(v.y * (float)mp[1]);
  xb[base + 2 * CHN] = (__bf16)(v.z * (float)mp[2]);
  xb[base + 3 * CHN] = (__bf16)(v.w * (float)mp[3]);
}

// ---------------- cnt[b] = sum(mask[b]) ----------------
__global__ void k_cnt(const uint8_t* __restrict__ mask, float* __restrict__ cnt) {
  __shared__ float sm[8];
  int b = blockIdx.x;
  float s = 0.f;
  for (int k = threadIdx.x; k < HWSZ; k += 256) s += (float)mask[b * HWSZ + k];
  s = waveReduce(s);
  if ((threadIdx.x & 31) == 0) sm[threadIdx.x >> 5] = s;
  __syncthreads();
  if (threadIdx.x == 0) {
    float t = 0.f;
    #pragma unroll
    for (int i = 0; i < 8; ++i) t += sm[i];
    cnt[b] = t;
  }
}

// ---------------- implicit-GEMM 3x3 conv, bf16 WMMA, f32 accum ----------------
// grid = B*H*2 workgroups; 256 threads = 8 waves; each WG: all 128 Co x 64 pos.
__global__ __launch_bounds__(256)
void k_conv(const __bf16* __restrict__ xin,   // NHWC bf16
            const __bf16* __restrict__ wt,    // [9][co][ci] bf16
            const float*  __restrict__ bias,
            const uint8_t* __restrict__ mask,
            float* __restrict__ out)          // NCHW f32
{
  __shared__ __align__(16) unsigned char smem[198 * 256];  // 3 rows x 66 cols x 128ch bf16

  int wg  = blockIdx.x;
  int wt2 = wg & 1;
  int h   = (wg >> 1) & (HT - 1);
  int b   = wg >> 8;
  int w0  = wt2 * 64;

  int t    = threadIdx.x;
  int lane = t & 31;
  int wave = t >> 5;            // co chunk
  int m16  = lane & 15;
  int half = lane >> 4;

  // Cooperative halo-tile load (zero pad out-of-range), XOR-swizzled 32B chunks.
  for (int idx = t; idx < 198 * 16; idx += 256) {
    int pos  = idx >> 4;
    int sub  = idx & 15;        // 16B unit within the 256B channel row
    int row  = pos / 66;
    int col  = pos - row * 66;
    int hy   = h - 1 + row;
    int wx   = w0 - 1 + col;
    uint4 val = {0u, 0u, 0u, 0u};
    if (hy >= 0 && hy < HT && wx >= 0 && wx < WD) {
      val = ((const uint4*)(xin + ((size_t)(b * HWSZ) + hy * WD + wx) * CHN))[sub];
    }
    int chunk = (sub >> 1) ^ (pos & 7);
    *(uint4*)(smem + pos * 256 + chunk * 32 + (sub & 1) * 16) = val;
  }
  __syncthreads();

  v8f acc[4] = {};

  for (int tap = 0; tap < 9; ++tap) {
    int ky = tap / 3;
    int kx = tap - ky * 3;
    const __bf16* wrow = wt + ((size_t)tap * CHN + wave * 16 + m16) * CHN;
    int prow = ky * 66 + kx;
    #pragma unroll
    for (int kk = 0; kk < 4; ++kk) {
      // A fragment: contiguous K order (lane half selects K 16..31)
      ABPack A;
      A.u[0] = *(const uint4*)(wrow + kk * 32 + half * 16);
      A.u[1] = *(const uint4*)(wrow + kk * 32 + half * 16 + 8);
      int cib = kk * 64 + half * 32;   // this lane's 32B of K within the 256B channel row

      // Issue all 4 independent B-fragment loads first so the DS latency is
      // pipelined against the WMMA issue (graduated s_wait_dscnt instead of 0x0).
      ABPack Bv[4];
      #pragma unroll
      for (int n = 0; n < 4; ++n) {
        int pos   = prow + n * 16 + m16;
        int chunk = (cib >> 5) ^ (pos & 7);
        const unsigned char* p = smem + pos * 256 + chunk * 32;
        Bv[n].u[0] = *(const uint4*)(p);
        Bv[n].u[1] = *(const uint4*)(p + 16);
      }
      #pragma unroll
      for (int n = 0; n < 4; ++n) {
        acc[n] = __builtin_amdgcn_wmma_f32_16x16x32_bf16(
            false, A.v, false, Bv[n].v, (short)0, acc[n], false, false);
      }
    }
  }

  // Store: C/D layout VGPR r -> rows (co) r and r+8; lanes = columns (pos).
  #pragma unroll
  for (int n = 0; n < 4; ++n) {
    int wx = w0 + n * 16 + m16;
    float mv = (float)mask[b * HWSZ + h * WD + wx];
    #pragma unroll
    for (int r = 0; r < 8; ++r) {
      int co = wave * 16 + r + 8 * half;
      out[((size_t)(b * CHN + co)) * HWSZ + h * WD + wx] = (acc[n][r] + bias[co]) * mv;
    }
  }
}

// ---------------- per-(b,g) sum / sumsq (input already masked) ----------------
__global__ void k_stats(const float* __restrict__ out, float* __restrict__ stats) {
  __shared__ float sm[16];
  int wg    = blockIdx.x;             // b*128 + g*8 + slice
  int slice = wg & 7;
  int g     = (wg >> 3) & (NGROUP - 1);
  int b     = wg >> 7;
  float s = 0.f, s2 = 0.f;
  for (int ci = 0; ci < CG; ++ci) {
    const float* p = out + ((size_t)(b * CHN + g * CG + ci)) * HWSZ + slice * (HWSZ / 8);
    for (int k = threadIdx.x; k < HWSZ / 8; k += 256) {
      float v = p[k];
      s += v; s2 += v * v;
    }
  }
  s = waveReduce(s); s2 = waveReduce(s2);
  if ((threadIdx.x & 31) == 0) { sm[threadIdx.x >> 5] = s; sm[8 + (threadIdx.x >> 5)] = s2; }
  __syncthreads();
  if (threadIdx.x == 0) {
    float a = 0.f, a2 = 0.f;
    #pragma unroll
    for (int i = 0; i < 8; ++i) { a += sm[i]; a2 += sm[8 + i]; }
    atomicAdd(&stats[(b * NGROUP + g) * 2], a);
    atomicAdd(&stats[(b * NGROUP + g) * 2 + 1], a2);
  }
}

// ---------------- norm1: affine + relu + mask -> bf16 NHWC (conv2 input) ----------------
__global__ void k_norm1(const float* __restrict__ out, const float* __restrict__ stats,
                        const float* __restrict__ cnt, const float* __restrict__ gamma,
                        const float* __restrict__ beta, const uint8_t* __restrict__ mask,
                        __bf16* __restrict__ xb) {
  int i = blockIdx.x * 256 + threadIdx.x;
  if (i >= BATCH * CHN * HWSZ / 4) return;
  int p4 = i & (HWSZ / 4 - 1);
  int c  = (i >> 12) & (CHN - 1);
  int b  = i >> 19;
  int g  = c >> 3;
  float cg   = fmaxf(cnt[b] * (float)CG, 1.f);
  float mean = stats[(b * NGROUP + g) * 2] / cg;
  float var  = stats[(b * NGROUP + g) * 2 + 1] / cg - mean * mean;
  float inv  = rsqrtf(var + EPSV);
  float ga = gamma[c], be = beta[c];
  float4 v = ((const float4*)out)[(size_t)i];
  const uint8_t* mp = mask + b * HWSZ + p4 * 4;
  int base = (b * HWSZ + p4 * 4) * CHN + c;
  float r;
  r = (v.x - mean) * inv * ga + be; xb[base]           = (__bf16)(fmaxf(r, 0.f) * (float)mp[0]);
  r = (v.y - mean) * inv * ga + be; xb[base + CHN]     = (__bf16)(fmaxf(r, 0.f) * (float)mp[1]);
  r = (v.z - mean) * inv * ga + be; xb[base + 2 * CHN] = (__bf16)(fmaxf(r, 0.f) * (float)mp[2]);
  r = (v.w - mean) * inv * ga + be; xb[base + 3 * CHN] = (__bf16)(fmaxf(r, 0.f) * (float)mp[3]);
}

// ---------------- norm2 + residual + relu + mask -> f32 NCHW output ----------------
__global__ void k_norm2(const float* __restrict__ out, const float* __restrict__ stats,
                        const float* __restrict__ cnt, const float* __restrict__ gamma,
                        const float* __restrict__ beta, const uint8_t* __restrict__ mask,
                        const float* __restrict__ x, float* __restrict__ dst) {
  int i = blockIdx.x * 256 + threadIdx.x;
  if (i >= BATCH * CHN * HWSZ / 4) return;
  int p4 = i & (HWSZ / 4 - 1);
  int c  = (i >> 12) & (CHN - 1);
  int b  = i >> 19;
  int g  = c >> 3;
  float cg   = fmaxf(cnt[b] * (float)CG, 1.f);
  float mean = stats[(b * NGROUP + g) * 2] / cg;
  float var  = stats[(b * NGROUP + g) * 2 + 1] / cg - mean * mean;
  float inv  = rsqrtf(var + EPSV);
  float ga = gamma[c], be = beta[c];
  float4 v  = ((const float4*)out)[(size_t)i];
  float4 xi = ((const float4*)x)[(size_t)i];
  const uint8_t* mp = mask + b * HWSZ + p4 * 4;
  float4 res;
  float m, r;
  m = (float)mp[0]; r = ((v.x - mean) * inv * ga + be) * m + xi.x * m; res.x = fmaxf(r, 0.f) * m;
  m = (float)mp[1]; r = ((v.y - mean) * inv * ga + be) * m + xi.y * m; res.y = fmaxf(r, 0.f) * m;
  m = (float)mp[2]; r = ((v.z - mean) * inv * ga + be) * m + xi.z * m; res.z = fmaxf(r, 0.f) * m;
  m = (float)mp[3]; r = ((v.w - mean) * inv * ga + be) * m + xi.w * m; res.w = fmaxf(r, 0.f) * m;
  ((float4*)dst)[(size_t)i] = res;
}

// ---------------- host-side pipeline ----------------
extern "C" void kernel_launch(void* const* d_in, const int* in_sizes, int n_in,
                              void* d_out, int out_size, void* d_ws, size_t ws_size,
                              hipStream_t stream) {
  const float*   x    = (const float*)d_in[0];
  const uint8_t* mask = (const uint8_t*)d_in[1];
  const float*   w1   = (const float*)d_in[2];
  const float*   b1   = (const float*)d_in[3];
  const float*   g1   = (const float*)d_in[4];
  const float*   bt1  = (const float*)d_in[5];
  const float*   w2   = (const float*)d_in[6];
  const float*   b2   = (const float*)d_in[7];
  const float*   g2   = (const float*)d_in[8];
  const float*   bt2  = (const float*)d_in[9];
  float* dst = (float*)d_out;

  // Workspace layout (~101.3 MB total)
  uint8_t* ws   = (uint8_t*)d_ws;
  __bf16* w1t   = (__bf16*)(ws);                    //   294,912 B
  __bf16* w2t   = (__bf16*)(ws + 294912);           //   294,912 B
  __bf16* xb    = (__bf16*)(ws + 589824);           // 33,554,432 B (bf16 NHWC staging)
  float*  obuf  = (float*) (ws + 34144256);         // 67,108,864 B (f32 conv output)
  float*  st1   = (float*) (ws + 101253120);        //     1,024 B
  float*  st2   = (float*) (ws + 101254144);        //     1,024 B
  float*  cnt   = (float*) (ws + 101255168);        //        32 B

  const int n4 = BATCH * CHN * HWSZ / 4;            // 4,194,304

  k_zero <<<3, 256, 0, stream>>>(st1, 520);         // st1 + st2 + cnt (contiguous)
  k_wconv<<<576, 256, 0, stream>>>(w1, w1t);
  k_wconv<<<576, 256, 0, stream>>>(w2, w2t);
  k_prep <<<n4 / 256, 256, 0, stream>>>(x, mask, xb);
  k_cnt  <<<BATCH, 256, 0, stream>>>(mask, cnt);

  k_conv <<<BATCH * HT * 2, 256, 0, stream>>>(xb, w1t, b1, mask, obuf);
  k_stats<<<BATCH * NGROUP * 8, 256, 0, stream>>>(obuf, st1);
  k_norm1<<<n4 / 256, 256, 0, stream>>>(obuf, st1, cnt, g1, bt1, mask, xb);

  k_conv <<<BATCH * HT * 2, 256, 0, stream>>>(xb, w2t, b2, mask, obuf);
  k_stats<<<BATCH * NGROUP * 8, 256, 0, stream>>>(obuf, st2);
  k_norm2<<<n4 / 256, 256, 0, stream>>>(obuf, st2, cnt, g2, bt2, mask, x, dst);
}
